// LSTMModule_29437705846950
// MI455X (gfx1250) — matrix operand order repaired and checked by reading, not compiled, not run on hardware
//
#include <hip/hip_runtime.h>
#include <hip/hip_bf16.h>

// ---------------------------------------------------------------------------
// 3-layer LSTM (B=128, T=1024, F=6, H=256) for gfx1250 (MI455X).
// Recurrent GEMM per step: gates[128,1024] = [x_t | h_{t-1}] (K=KX+256) @ W^T
// done with v_wmma_f32_16x16x32_bf16. 16 persistent workgroups, one grid
// barrier per timestep (h_{t-1} is read from hs[t-1], h_t written to hs[t],
// so reads/writes never alias within a step).
// ---------------------------------------------------------------------------

typedef __attribute__((ext_vector_type(16))) __bf16 v16bf;
typedef __attribute__((ext_vector_type(8)))  float  v8f;

#define T_STEPS 1024
#define BATCH   128
#define HID     256
#define GATES   1024   // 4*HID
#define NWG     16     // persistent workgroups in recurrent kernel
#define THREADS 256    // 8 waves of 32

// ---------------- small helpers ----------------
__device__ __forceinline__ unsigned short f2bf(float f) {
  unsigned u = __float_as_uint(f);
  unsigned r = u + 0x7FFFu + ((u >> 16) & 1u);   // round-to-nearest-even
  return (unsigned short)(r >> 16);
}
__device__ __forceinline__ float bf2f(unsigned short b) {
  return __uint_as_float(((unsigned)b) << 16);
}
__device__ __forceinline__ float sigm(float x) {
  return 1.0f / (1.0f + __expf(-x));
}
__device__ __forceinline__ float tanh_fast(float x) {
  float e = __expf(2.0f * x);
  return 1.0f - 2.0f / (e + 1.0f);   // exact at both saturation ends
}

// monotonic grid barrier across NWG persistent workgroups
__device__ __forceinline__ void grid_barrier(unsigned int* counter, unsigned target) {
  __syncthreads();
  if (threadIdx.x == 0) {
    __hip_atomic_fetch_add(counter, 1u, __ATOMIC_ACQ_REL, __HIP_MEMORY_SCOPE_AGENT);
    while (__hip_atomic_load(counter, __ATOMIC_ACQUIRE, __HIP_MEMORY_SCOPE_AGENT) < target) {
      __builtin_amdgcn_s_sleep(1);
    }
  }
  __syncthreads();
}

// ---------------- WMMA inner step ----------------
// A fragment (16x32 bf16) from LDS per ISA layout:
//   lane: M = mtile*16 + (lane&15), half = lane>>4
//   VGPR v (pair pos p): K = 2v + (v>=4 ? 8 : 0) + half*8 + p
// B fragment: pre-packed contiguous 32B per lane in wpack.
__device__ __forceinline__ void mma_kt(const unsigned short* lds, int kchunk,
                                       int ktloc, int ktglob,
                                       int mtile, int wg, int half, int ln, int lane,
                                       int KT, const unsigned short* __restrict__ wpack,
                                       v8f& g0, v8f& g1, v8f& g2, v8f& g3)
{
  union AU { v16bf v; unsigned int u[8]; } a;
  const unsigned short* ap = lds + (mtile * 16 + ln) * kchunk + ktloc * 32 + half * 8;
  a.u[0] = *(const unsigned int*)(ap +  0);
  a.u[1] = *(const unsigned int*)(ap +  2);
  a.u[2] = *(const unsigned int*)(ap +  4);
  a.u[3] = *(const unsigned int*)(ap +  6);
  a.u[4] = *(const unsigned int*)(ap + 16);
  a.u[5] = *(const unsigned int*)(ap + 18);
  a.u[6] = *(const unsigned int*)(ap + 20);
  a.u[7] = *(const unsigned int*)(ap + 22);

  union BU { v16bf v; uint4 q[2]; } b;
#pragma unroll
  for (int g = 0; g < 4; ++g) {
    const int ntile = g * 16 + wg;                      // N tile: gate g, this WG's 16 hidden cols
    const uint4* bp = (const uint4*)(wpack + (((long)(ntile * KT + ktglob)) * 32 + lane) * 16);
    b.q[0] = bp[0];
    b.q[1] = bp[1];
    v8f& acc = (g == 0) ? g0 : (g == 1) ? g1 : (g == 2) ? g2 : g3;
    acc = __builtin_amdgcn_wmma_f32_16x16x32_bf16(false, a.v, false, b.v,
                                                  (short)0, acc, false, false);
  }
}

// ---------------- prologue kernels ----------------

// Pack [w_ih | w_hh] (fp32) into bf16 WMMA B-fragment order:
//   dst[((ntile*KT + kt)*32 + lane)*16 + e]  where
//   n = ntile*16 + (lane&15),  k = kt*32 + (e&7) + (lane>>4)*8 + (e>>3)*16
__global__ void k_pack_w(const float* __restrict__ wih0,
                         const float* __restrict__ wih12,
                         const float* __restrict__ whh,
                         unsigned short* __restrict__ dst,
                         int layer, int KT, int KX)
{
  long idx = (long)blockIdx.x * blockDim.x + threadIdx.x;
  long total = 64L * KT * 512;
  if (idx >= total) return;
  int e    = (int)(idx & 15);
  int lane = (int)((idx >> 4) & 31);
  long rem = idx >> 9;            // ntile*KT + kt
  int kt    = (int)(rem % KT);
  int ntile = (int)(rem / KT);
  int half = lane >> 4, ln = lane & 15;
  int n = ntile * 16 + ln;
  int v = e >> 1, pos = e & 1;
  int k = kt * 32 + 2 * (v & 3) + pos + half * 8 + (v >> 2) * 16;

  float val = 0.0f;
  if (k < KX) {                                   // input-weight part
    if (layer == 0) { if (k < 6) val = wih0[n * 6 + k]; }
    else            { val = wih12[((long)(layer - 1) * GATES + n) * HID + k]; }
  } else {                                        // recurrent-weight part
    int kh = k - KX;
    val = whh[((long)layer * GATES + n) * HID + kh];
  }
  dst[idx] = f2bf(val);
}

// signal [B,T,F] fp32 -> x0 bf16 [T][B][32] (F=6 zero-padded to 32)
__global__ void k_pack_x0(const float* __restrict__ sig, unsigned short* __restrict__ dst)
{
  int idx = blockIdx.x * blockDim.x + threadIdx.x;
  if (idx >= T_STEPS * BATCH * 32) return;
  int kk = idx & 31;
  int b  = (idx >> 5) & (BATCH - 1);
  int t  = idx >> 12;
  float v = (kk < 6) ? sig[(b * T_STEPS + t) * 6 + kk] : 0.0f;
  dst[idx] = f2bf(v);
}

// combined bias b_ih + b_hh, plus zero the barrier counters
__global__ void k_misc(const float* __restrict__ bih, const float* __restrict__ bhh,
                       float* __restrict__ bias, unsigned int* __restrict__ cnt)
{
  int idx = blockIdx.x * blockDim.x + threadIdx.x;
  if (idx < 3 * GATES) bias[idx] = bih[idx] + bhh[idx];
  if (idx < 8) cnt[idx] = 0u;
}

// ---------------- recurrent LSTM layer (persistent, WMMA) ----------------
__global__ void __launch_bounds__(THREADS, 1)
k_lstm(const unsigned short* __restrict__ xin,    // [T][128][KX] bf16
       const unsigned short* __restrict__ wpack,  // packed B fragments
       const float* __restrict__ bias,            // [1024]
       unsigned short* __restrict__ hs,           // [T][128][256] bf16 (output + h state)
       unsigned int* __restrict__ counter,
       int KX, int KT)
{
  __shared__ unsigned short lds_a[BATCH * HID];   // 64 KB staging for A chunks

  const int wg    = blockIdx.x;          // hidden cols [wg*16, wg*16+16)
  const int wave  = threadIdx.x >> 5;
  const int lane  = threadIdx.x & 31;
  const int half  = lane >> 4;
  const int ln    = lane & 15;
  const int mtile = wave;                // batch rows [mtile*16, mtile*16+16)
  const int KXT   = KX >> 5;             // K-tiles in input chunk

  const float bi = bias[0 * HID + wg * 16 + ln];
  const float bf = bias[1 * HID + wg * 16 + ln];
  const float bg = bias[2 * HID + wg * 16 + ln];
  const float bo = bias[3 * HID + wg * 16 + ln];

  v8f cst = {0.f, 0.f, 0.f, 0.f, 0.f, 0.f, 0.f, 0.f};   // cell state, resident in VGPRs
  unsigned bar = 0;

  for (int t = 0; t < T_STEPS; ++t) {
    // ---- stage x_t chunk into LDS (contiguous block copy) ----
    {
      const uint4* s = (const uint4*)(xin + (long)t * BATCH * KX);
      uint4* d = (uint4*)lds_a;
      const int n16 = (BATCH * KX) >> 3;
      for (int i = threadIdx.x; i < n16; i += THREADS) d[i] = s[i];
      if (t + 1 < T_STEPS)
        __builtin_prefetch(xin + (long)(t + 1) * BATCH * KX, 0, 0);
    }
    __syncthreads();

    v8f g0 = {0.f,0.f,0.f,0.f,0.f,0.f,0.f,0.f};
    v8f g1 = {0.f,0.f,0.f,0.f,0.f,0.f,0.f,0.f};
    v8f g2 = {0.f,0.f,0.f,0.f,0.f,0.f,0.f,0.f};
    v8f g3 = {0.f,0.f,0.f,0.f,0.f,0.f,0.f,0.f};

    for (int kt = 0; kt < KXT; ++kt)
      mma_kt(lds_a, KX, kt, kt, mtile, wg, half, ln, lane, KT, wpack, g0, g1, g2, g3);
    __syncthreads();

    // ---- stage h_{t-1} chunk into LDS (zeros at t==0) ----
    {
      uint4* d = (uint4*)lds_a;
      const int n16 = (BATCH * HID) >> 3;
      if (t == 0) {
        const uint4 z = {0u, 0u, 0u, 0u};
        for (int i = threadIdx.x; i < n16; i += THREADS) d[i] = z;
      } else {
        const uint4* s = (const uint4*)(hs + (long)(t - 1) * BATCH * HID);
        for (int i = threadIdx.x; i < n16; i += THREADS) d[i] = s[i];
      }
    }
    __syncthreads();

#pragma unroll 2
    for (int kt = 0; kt < HID / 32; ++kt)
      mma_kt(lds_a, HID, kt, KXT + kt, mtile, wg, half, ln, lane, KT, wpack, g0, g1, g2, g3);

    // ---- cell update (wave-local: rows mtile*16+half*8+j, col wg*16+ln) ----
#pragma unroll
    for (int j = 0; j < 8; ++j) {
      float iv = sigm(g0[j] + bi);
      float fv = sigm(g1[j] + bf);
      float gv = tanh_fast(g2[j] + bg);
      float ov = sigm(g3[j] + bo);
      float cn = fv * cst[j] + iv * gv;
      cst[j] = cn;
      float hv = ov * tanh_fast(cn);
      int row = mtile * 16 + half * 8 + j;
      int col = wg * 16 + ln;
      hs[((long)t * BATCH + row) * HID + col] = f2bf(hv);
    }
    __threadfence();
    grid_barrier(counter, (++bar) * NWG);   // hs[t] complete before step t+1 reads it
  }
}

// ---------------- classifier epilogue ----------------
__global__ void k_cls(const unsigned short* __restrict__ h,   // hs[T-1] of last layer
                      const float* __restrict__ wcls,
                      const float* __restrict__ bcls,
                      float* __restrict__ out)
{
  int idx = blockIdx.x * blockDim.x + threadIdx.x;
  if (idx >= BATCH * 10) return;
  int b = idx / 10, c = idx % 10;
  float s = bcls[c];
  for (int k = 0; k < HID; ++k) {
    float hv = sigm(bf2f(h[b * HID + k]));
    s += hv * wcls[c * HID + k];
  }
  out[idx] = s;
}

// ---------------- host launcher ----------------
extern "C" void kernel_launch(void* const* d_in, const int* in_sizes, int n_in,
                              void* d_out, int out_size, void* d_ws, size_t ws_size,
                              hipStream_t stream) {
  const float* sig   = (const float*)d_in[0];
  const float* wih0  = (const float*)d_in[1];
  const float* wih12 = (const float*)d_in[2];
  const float* whh   = (const float*)d_in[3];
  const float* bih   = (const float*)d_in[4];
  const float* bhh   = (const float*)d_in[5];
  const float* wcls  = (const float*)d_in[6];
  const float* bcls  = (const float*)d_in[7];
  float* out = (float*)d_out;

  char* ws = (char*)d_ws;
  size_t off = 0;
  auto carve = [&](size_t bytes) -> char* {
    off = (off + 255) & ~(size_t)255;
    char* p = ws + off;
    off += bytes;
    return p;
  };
  const size_t WPACK_ELEMS = 64UL * 16 * 32 * 16;                 // per-layer stride (KT<=16)
  unsigned short* wpack = (unsigned short*)carve(3 * WPACK_ELEMS * 2);
  unsigned short* x0    = (unsigned short*)carve((size_t)T_STEPS * BATCH * 32 * 2);
  unsigned short* hsA   = (unsigned short*)carve((size_t)T_STEPS * BATCH * HID * 2);
  unsigned short* hsB   = (unsigned short*)carve((size_t)T_STEPS * BATCH * HID * 2);
  float*          bias  = (float*)carve(3 * GATES * 4);
  unsigned int*   cnt   = (unsigned int*)carve(64);

  const int KTs[3] = {9, 16, 16};    // layer0: K = 32(x-pad) + 256(h) = 288 -> 9 K-tiles
  const int KXs[3] = {32, 256, 256};

  for (int l = 0; l < 3; ++l) {
    long total = 64L * KTs[l] * 512;
    k_pack_w<<<(unsigned)((total + 255) / 256), 256, 0, stream>>>(
        wih0, wih12, whh, wpack + (size_t)l * WPACK_ELEMS, l, KTs[l], KXs[l]);
  }
  k_pack_x0<<<(T_STEPS * BATCH * 32) / 256, 256, 0, stream>>>(sig, x0);
  k_misc<<<(3 * GATES + 255) / 256, 256, 0, stream>>>(bih, bhh, bias, cnt);

  const unsigned short* xins[3] = {x0, hsA, hsB};
  unsigned short*       houts[3] = {hsA, hsB, hsA};
  for (int l = 0; l < 3; ++l) {
    k_lstm<<<NWG, THREADS, 0, stream>>>(xins[l], wpack + (size_t)l * WPACK_ELEMS,
                                        bias + l * GATES, houts[l],
                                        cnt + l, KXs[l], KTs[l]);
  }
  // final hidden state of last layer = hs[T-1] of layer 2 (stored in hsA)
  k_cls<<<(BATCH * 10 + 255) / 256, 256, 0, stream>>>(
      hsA + (size_t)(T_STEPS - 1) * BATCH * HID, wcls, bcls, out);
}
